// myRNN_489626272013
// MI455X (gfx1250) — compile-verified, matrix operand order checked by reading, not run on hardware
//
#include <hip/hip_runtime.h>
#include <hip/hip_fp16.h>

typedef __attribute__((ext_vector_type(16))) _Float16 v16h;
typedef __attribute__((ext_vector_type(8)))  float    v8f;

#define HDIM 256
#define NTOK (16 * 4096)
#define LDS_STRIDE 264            // halves per row: 256 + 8 pad -> conflict-light b128 reads

// workspace layout (bytes)
#define WT_OFF   0                              // _Float16 [2][256][256] transposed f16 weights
#define WR2_OFF  (2 * 256 * 256 * 2)            // float [256][2]  (sigmoid(b1[3]) folded into Wr)
#define B1_OFF   (WR2_OFF + 256 * 2 * 4)        // float [256]  layer0 input-gate bias
#define B2_OFF   (B1_OFF + 256 * 4)             // float [256]  layer0 candidate bias

__device__ __forceinline__ float sigm_f(float x) {
    return 1.0f / (1.0f + __expf(-x));
}
__device__ __forceinline__ float tanh_f(float x) {
    float e = __expf(2.0f * x);      // inf -> 1, 0 -> -1 : correct saturation
    return 1.0f - 2.0f / (e + 1.0f);
}

// ---------------------------------------------------------------------------
// Prep: Wt[g][j][k] = (f16) W[0][g+1][k][j]  (g=0 -> input gate, g=1 -> candidate)
//       wr2[j][o]   = sigmoid(b[1][3][j]) * Wr[j][o]
//       b1[j], b2[j] = layer0 biases for gates 1,2
// ---------------------------------------------------------------------------
__global__ void prep_kernel(const float* __restrict__ W,
                            const float* __restrict__ b,
                            const float* __restrict__ Wr,
                            void* __restrict__ ws) {
    _Float16* wt = (_Float16*)((char*)ws + WT_OFF);
    float* wr2   = (float*)((char*)ws + WR2_OFF);
    float* b1    = (float*)((char*)ws + B1_OFF);
    float* b2    = (float*)((char*)ws + B2_OFF);

    int blk = blockIdx.x;
    int tid = threadIdx.x;
    if (blk < 512) {
        int idx = blk * 256 + tid;          // 0 .. 131071
        int g = idx >> 16;                  // 0,1
        int j = (idx >> 8) & 255;           // output channel
        int k = idx & 255;                  // input feature (only first D rows matter)
        // W flat (L,4,512,256): ((0*4 + (g+1))*512 + k)*256 + j
        float w = W[(((size_t)(g + 1)) * 512 + k) * 256 + j];
        wt[((size_t)(g * 256 + j)) * 256 + k] = (_Float16)w;
    } else {
        int j = tid;
        float o1 = sigm_f(b[(1 * 4 + 3) * 256 + j]);   // layer1 output gate = pure bias
        wr2[j * 2 + 0] = o1 * Wr[j * 2 + 0];
        wr2[j * 2 + 1] = o1 * Wr[j * 2 + 1];
        b1[j] = b[256 + j];                             // b[0][1][j]
        b2[j] = b[512 + j];                             // b[0][2][j]
    }
}

// ---------------------------------------------------------------------------
// Main: per wave, 16 tokens. g1 = x@W1t^T + b1, g2 = x@W2t^T + b2 via
// v_wmma_f32_16x16x32_f16; h = tanh(tanh(g2)*sigmoid(g1)); out = h @ Wr' + br.
// ---------------------------------------------------------------------------
__global__ void __launch_bounds__(256)
lstm_wmma_kernel(const float* __restrict__ x,
                 const float* __restrict__ br,
                 const void* __restrict__ ws,
                 float* __restrict__ out) {
    extern __shared__ _Float16 lds_w[];   // [2][256][LDS_STRIDE]

    const _Float16* wt  = (const _Float16*)((const char*)ws + WT_OFF);
    const float*    wr2 = (const float*)((const char*)ws + WR2_OFF);
    const float*    bs1 = (const float*)((const char*)ws + B1_OFF);
    const float*    bs2 = (const float*)((const char*)ws + B2_OFF);

    // Cooperative copy of both f16 weight matrices into padded LDS.
    // 512 rows x 32 uint4 per row = 16384 chunks / 256 threads.
    for (int c = threadIdx.x; c < 512 * 32; c += 256) {
        int row = c >> 5;
        int kc  = c & 31;
        uint4 v = ((const uint4*)(wt + (size_t)row * 256))[kc];
        ((uint4*)(lds_w + (size_t)row * LDS_STRIDE))[kc] = v;
    }
    __syncthreads();

    const int lane  = threadIdx.x & 31;
    const int wid   = threadIdx.x >> 5;
    const int tbase = (blockIdx.x * 8 + wid) * 16;
    const int mrow  = lane & 15;            // row (token) / column (channel) within tile
    const int kofs  = (lane >> 4) * 8;      // half-wave k offset per ISA 16-bit layout

    // ---- Load A fragments: 16 tokens x 256 features, f32 -> f16 in registers.
    // v16h halves: [0..7] = k0..k0+7, [8..15] = k0+16..k0+23 (k0 = kk*32 + kofs)
    v16h A[8];
    const float* xrow = x + ((size_t)(tbase + mrow)) * 256;
#pragma unroll
    for (int kk = 0; kk < 8; ++kk) {
        const float* p = xrow + kk * 32 + kofs;
        float4 f0 = ((const float4*)p)[0];
        float4 f1 = ((const float4*)p)[1];
        float4 f2 = ((const float4*)(p + 16))[0];
        float4 f3 = ((const float4*)(p + 16))[1];
        v16h a;
        a[0]  = (_Float16)f0.x; a[1]  = (_Float16)f0.y;
        a[2]  = (_Float16)f0.z; a[3]  = (_Float16)f0.w;
        a[4]  = (_Float16)f1.x; a[5]  = (_Float16)f1.y;
        a[6]  = (_Float16)f1.z; a[7]  = (_Float16)f1.w;
        a[8]  = (_Float16)f2.x; a[9]  = (_Float16)f2.y;
        a[10] = (_Float16)f2.z; a[11] = (_Float16)f2.w;
        a[12] = (_Float16)f3.x; a[13] = (_Float16)f3.y;
        a[14] = (_Float16)f3.z; a[15] = (_Float16)f3.w;
        A[kk] = a;
    }

    float p0[8] = {0.f, 0.f, 0.f, 0.f, 0.f, 0.f, 0.f, 0.f};
    float p1[8] = {0.f, 0.f, 0.f, 0.f, 0.f, 0.f, 0.f, 0.f};

    // 16 column tiles in 4 chunks of 4 tiles; both gate matrices per chunk.
    for (int ch = 0; ch < 4; ++ch) {
        v8f acc1[4] = {};
        v8f acc2[4] = {};
#pragma unroll
        for (int kk = 0; kk < 8; ++kk) {
#pragma unroll
            for (int t = 0; t < 4; ++t) {
                const int j = (ch * 4 + t) * 16 + mrow;   // B column = lane's channel
                const _Float16* base1 = lds_w + (size_t)(0 * 256 + j) * LDS_STRIDE + kk * 32 + kofs;
                const _Float16* base2 = lds_w + (size_t)(1 * 256 + j) * LDS_STRIDE + kk * 32 + kofs;
                union { v16h v; uint4 u[2]; } bf1, bf2;
                bf1.u[0] = *(const uint4*)base1;
                bf1.u[1] = *(const uint4*)(base1 + 16);
                bf2.u[0] = *(const uint4*)base2;
                bf2.u[1] = *(const uint4*)(base2 + 16);
                acc1[t] = __builtin_amdgcn_wmma_f32_16x16x32_f16(
                    false, A[kk], false, bf1.v, (short)0, acc1[t], false, false);
                acc2[t] = __builtin_amdgcn_wmma_f32_16x16x32_f16(
                    false, A[kk], false, bf2.v, (short)0, acc2[t], false, false);
            }
        }
        // Elementwise gates + weighted partial sums.
        // Lane holds channel j; acc[v] is token row (v + 8*(lane>=16)).
#pragma unroll
        for (int t = 0; t < 4; ++t) {
            const int j = (ch * 4 + t) * 16 + mrow;
            float b1j = bs1[j], b2j = bs2[j];
            float w0 = wr2[j * 2 + 0], w1 = wr2[j * 2 + 1];
#pragma unroll
            for (int v = 0; v < 8; ++v) {
                float g1 = acc1[t][v] + b1j;
                float g2 = acc2[t][v] + b2j;
                float c0 = tanh_f(g2) * sigm_f(g1);   // layer0 new_c
                float h  = tanh_f(c0);                // layer1 tanh(old c); o1 folded in wr2
                p0[v] += h * w0;
                p1[v] += h * w1;
            }
        }
    }

    // Reduce over 16 channels per lane-group (masks <=8 stay within a 16-lane half).
#pragma unroll
    for (int m = 1; m <= 8; m <<= 1) {
#pragma unroll
        for (int v = 0; v < 8; ++v) {
            p0[v] += __shfl_xor(p0[v], m, 32);
            p1[v] += __shfl_xor(p1[v], m, 32);
        }
    }

    if ((lane & 15) == 0) {
        const float br0 = br[0], br1 = br[1];
        const int tok = tbase + (lane >> 4) * 8;   // lane0 -> rows 0..7, lane16 -> rows 8..15
#pragma unroll
        for (int v = 0; v < 8; ++v) {
            float2 o = make_float2(p0[v] + br0, p1[v] + br1);
            *(float2*)(out + (size_t)(tok + v) * 2) = o;
        }
    }
}

extern "C" void kernel_launch(void* const* d_in, const int* in_sizes, int n_in,
                              void* d_out, int out_size, void* d_ws, size_t ws_size,
                              hipStream_t stream) {
    const float* x  = (const float*)d_in[0];
    const float* W  = (const float*)d_in[1];
    const float* b  = (const float*)d_in[2];
    const float* Wr = (const float*)d_in[3];
    const float* br = (const float*)d_in[4];
    float* out = (float*)d_out;

    // Stage transposed f16 weights + fused regressor into workspace.
    prep_kernel<<<513, 256, 0, stream>>>(W, b, Wr, d_ws);

    // 65536 tokens / (8 waves * 16 tokens) = 512 workgroups.
    size_t shmem = (size_t)2 * 256 * LDS_STRIDE * sizeof(_Float16);   // 270336 B < 320 KB/WGP
    lstm_wmma_kernel<<<512, 256, shmem, stream>>>(x, br, d_ws, out);
}
